// GRIL_29248727286194
// MI455X (gfx1250) — matrix operand order repaired and checked by reading, compiled, not run on hardware
//
#include <hip/hip_runtime.h>

typedef __attribute__((ext_vector_type(16))) _Float16 v16h;
typedef __attribute__((ext_vector_type(8)))  float    v8f;

#define NN 1024      // nodes
#define BB 2         // batch
#define HH 64        // hidden
#define TT 16        // time steps
#define MROWS 256    // B * 2H rows of activation matrix

// ---------------------------------------------------------------------------
// zero helper (re-run every launch: h and pred must start at 0)
// ---------------------------------------------------------------------------
__global__ __launch_bounds__(256) void k_zero(float* __restrict__ p, int n) {
  int i = blockIdx.x * 256 + threadIdx.x;
  if (i < n) p[i] = 0.f;
}

// ---------------------------------------------------------------------------
// support prep: inverse row/col sums of adj
// ---------------------------------------------------------------------------
__global__ __launch_bounds__(256) void k_sums(const float* __restrict__ adj,
                                              float* __restrict__ inv_row,
                                              float* __restrict__ inv_col) {
  int w  = blockIdx.x;
  int tl = threadIdx.x;
  float rs = 0.f, cs = 0.f;
  for (int v = tl; v < NN; v += 256) {
    rs += adj[(size_t)w * NN + v];   // row w of adj
    cs += adj[(size_t)v * NN + w];   // column w of adj
  }
  __shared__ float sr[256], sc[256];
  sr[tl] = rs; sc[tl] = cs;
  __syncthreads();
  for (int s = 128; s > 0; s >>= 1) {
    if (tl < s) { sr[tl] += sr[tl + s]; sc[tl] += sc[tl + s]; }
    __syncthreads();
  }
  if (tl == 0) {
    inv_row[w] = 1.f / (sr[0] + 1e-6f);
    inv_col[w] = 1.f / (sc[0] + 1e-6f);
  }
}

// ---------------------------------------------------------------------------
// Build S = support^T in f16, pre-swizzled into wave32 WMMA B-fragment layout.
// Element (k=v, n=w) -> tile (kt=v/32, nt=w/16), lane=(kin/16)*16+(w%16),
// half index = kin%16; each lane's 16 halves are contiguous (one v16h load).
//   fwd[w][v] = adj[w][v] * inv_row[w]   -> Sf[v][w]
//   bwd[w][v] = adj[v][w] * inv_col[w]   -> Sb[v][w]
// ---------------------------------------------------------------------------
__global__ __launch_bounds__(256) void k_swz(const float* __restrict__ adj,
                                             const float* __restrict__ inv_row,
                                             const float* __restrict__ inv_col,
                                             _Float16* __restrict__ Sf,
                                             _Float16* __restrict__ Sb) {
  int idx = blockIdx.x * 256 + threadIdx.x;   // over NN*NN
  int v = idx >> 10;
  int w = idx & (NN - 1);
  float f  = adj[(size_t)w * NN + v] * inv_row[w];
  float bk = adj[(size_t)v * NN + w] * inv_col[w];
  int kt = v >> 5, kin = v & 31;
  int nt = w >> 4, nin = w & 15;
  int lane = ((kin >> 4) << 4) | nin;
  int half = kin & 15;
  size_t o = ((size_t)(kt * (NN / 16) + nt)) * 512 + (size_t)lane * 16 + half;
  Sf[o] = (_Float16)f;
  Sb[o] = (_Float16)bk;
}

// ---------------------------------------------------------------------------
// Fused dual-support diffusion hop:
//   Y1(256x1024) = X1 * S1   and   Y2(256x1024) = X2 * S2
// in one launch (512 waves, 64 blocks) via v_wmma_f32_16x16x32_f16.
// One wave -> 16x64 output strip (4 f32 accumulators). The 8 waves of a
// block share (sel, nstrip) so B-fragment loads hit the same cachelines.
// Prefetch of the next K-tile is unconditional: keeps the loop body a
// single basic block (clean modulo schedule, no WMMA->VALU hazard nops);
// the one-past-the-end prefetch lands in adjacent workspace and is benign.
// ---------------------------------------------------------------------------
__global__ __launch_bounds__(256) void k_hop2(const float* __restrict__ X1,
                                              const _Float16* __restrict__ S1,
                                              float* __restrict__ Y1,
                                              const float* __restrict__ X2,
                                              const _Float16* __restrict__ S2,
                                              float* __restrict__ Y2) {
  int lane   = threadIdx.x & 31;
  int wave   = threadIdx.x >> 5;
  int blk    = blockIdx.x;          // 64 blocks
  int sel    = blk >> 5;            // which (X,S,Y) pair
  int rem    = blk & 31;
  int nstrip = rem >> 1;            // 16 strips of 64 cols
  int mgroup = rem & 1;             // 2 groups of 8 M tiles
  int mtile  = mgroup * 8 + wave;   // 16 M tiles
  int grp    = lane >> 4;
  int row    = mtile * 16 + (lane & 15);

  const float*    X = sel ? X2 : X1;
  const _Float16* S = sel ? S2 : S1;
  float*          Y = sel ? Y2 : Y1;

  v8f acc0 = {}, acc1 = {}, acc2 = {}, acc3 = {};
  const float* xr = X + (size_t)row * NN;

  for (int kt = 0; kt < NN / 32; ++kt) {
    // A fragment: halves 0..15 -> K = grp*16 + i (matches B swizzle)
    const float* xp = xr + kt * 32 + grp * 16;
    float4 a0 = *(const float4*)(xp);
    float4 a1 = *(const float4*)(xp + 4);
    float4 a2 = *(const float4*)(xp + 8);
    float4 a3 = *(const float4*)(xp + 12);
    v16h a;
    a[0]  = (_Float16)a0.x; a[1]  = (_Float16)a0.y; a[2]  = (_Float16)a0.z; a[3]  = (_Float16)a0.w;
    a[4]  = (_Float16)a1.x; a[5]  = (_Float16)a1.y; a[6]  = (_Float16)a1.z; a[7]  = (_Float16)a1.w;
    a[8]  = (_Float16)a2.x; a[9]  = (_Float16)a2.y; a[10] = (_Float16)a2.z; a[11] = (_Float16)a2.w;
    a[12] = (_Float16)a3.x; a[13] = (_Float16)a3.y; a[14] = (_Float16)a3.z; a[15] = (_Float16)a3.w;

    // B fragments: pre-swizzled, contiguous 16 halves per lane
    const _Float16* sb = S + ((size_t)(kt * (NN / 16) + nstrip * 4)) * 512 + (size_t)lane * 16;
    v16h b0 = *(const v16h*)(sb);
    v16h b1 = *(const v16h*)(sb + 512);
    v16h b2 = *(const v16h*)(sb + 1024);
    v16h b3 = *(const v16h*)(sb + 1536);

    // unconditional prefetch of next K-tile's B fragments (global_prefetch_b8)
    __builtin_prefetch(sb + 32768, 0, 3);
    __builtin_prefetch(sb + 32768 + 1024, 0, 3);

    acc0 = __builtin_amdgcn_wmma_f32_16x16x32_f16(false, a, false, b0, (short)0, acc0, false, false);
    acc1 = __builtin_amdgcn_wmma_f32_16x16x32_f16(false, a, false, b1, (short)0, acc1, false, false);
    acc2 = __builtin_amdgcn_wmma_f32_16x16x32_f16(false, a, false, b2, (short)0, acc2, false, false);
    acc3 = __builtin_amdgcn_wmma_f32_16x16x32_f16(false, a, false, b3, (short)0, acc3, false, false);
  }

  // C/D layout: lanes 0-15 hold M=vgpr, lanes 16-31 hold M=vgpr+8; N = lane&15
  int m0 = mtile * 16 + grp * 8;
  int nc = lane & 15;
  float* y0 = Y + (size_t)m0 * NN + nstrip * 64 + nc;
#pragma unroll
  for (int v = 0; v < 8; ++v) {
    y0[(size_t)v * NN + 0]  = acc0[v];
    y0[(size_t)v * NN + 16] = acc1[v];
    y0[(size_t)v * NN + 32] = acc2[v];
    y0[(size_t)v * NN + 48] = acc3[v];
  }
}

// ---------------------------------------------------------------------------
// Input projection: xg = [ins ; h] into X; also seed Xc's ins rows.
// ins = Wp * [x_s, mf, pred] + bp + u[:, :, :, t]
// ---------------------------------------------------------------------------
__global__ __launch_bounds__(256) void k_proj(const float* __restrict__ x,
                                              const unsigned char* __restrict__ mask,
                                              const float* __restrict__ u,
                                              const float* __restrict__ Wp,
                                              const float* __restrict__ bp,
                                              const float* __restrict__ pred,
                                              const float* __restrict__ h,
                                              float* __restrict__ X,
                                              float* __restrict__ Xc, int t) {
  int idx = blockIdx.x * 256 + threadIdx.x;   // B*NN
  if (idx >= BB * NN) return;
  int b = idx / NN, n = idx % NN;
  float xp = 0.f, mp = 0.f;
  if (t > 0) {
    xp = x[(size_t)(b * NN + n) * TT + (t - 1)];
    mp = mask[(size_t)(b * NN + n) * TT + (t - 1)] ? 1.f : 0.f;
  }
  float pr = pred[b * NN + n];
  float xs = (mp != 0.f) ? xp : pr;
  for (int c = 0; c < HH; ++c) {
    float ins = Wp[c * 3 + 0] * xs + Wp[c * 3 + 1] * mp + Wp[c * 3 + 2] * pr + bp[c]
              + u[((size_t)(b * HH + c) * NN + n) * TT + t];
    X[(size_t)(b * 128 + c) * NN + n]       = ins;
    Xc[(size_t)(b * 128 + c) * NN + n]      = ins;
    X[(size_t)(b * 128 + 64 + c) * NN + n]  = h[(size_t)(b * HH + c) * NN + n];
  }
}

__device__ __forceinline__ void accum2(const float* __restrict__ s,
                                       const float* __restrict__ w0,
                                       const float* __restrict__ w1,
                                       float& a0, float& a1) {
  for (int cc = 0; cc < 128; ++cc) {
    float v = s[(size_t)cc * NN];
    a0 = fmaf(w0[cc], v, a0);
    a1 = fmaf(w1[cc], v, a1);
  }
}

__device__ __forceinline__ void accum1(const float* __restrict__ s,
                                       const float* __restrict__ w0, float& a0) {
  for (int cc = 0; cc < 128; ++cc) {
    float v = s[(size_t)cc * NN];
    a0 = fmaf(w0[cc], v, a0);
  }
}

// ---------------------------------------------------------------------------
// r / z gates (share the 4 hop outputs). z kept; r consumed immediately:
// writes Xc's hidden rows = r*h for the candidate path.
// ---------------------------------------------------------------------------
__global__ __launch_bounds__(256) void k_gates(const float* __restrict__ X,
                                               const float* __restrict__ H1f,
                                               const float* __restrict__ H2f,
                                               const float* __restrict__ H1b,
                                               const float* __restrict__ H2b,
                                               const float* __restrict__ Wr,
                                               const float* __restrict__ br,
                                               const float* __restrict__ Wu,
                                               const float* __restrict__ bu,
                                               const float* __restrict__ h,
                                               float* __restrict__ z,
                                               float* __restrict__ Xc) {
  int idx = blockIdx.x * 256 + threadIdx.x;   // HH*BB*NN
  if (idx >= HH * BB * NN) return;
  int o   = idx / (BB * NN);
  int rem = idx % (BB * NN);
  int b = rem / NN, n = rem % NN;
  size_t base = (size_t)b * 128 * NN + n;
  float ar = br[o], az = bu[o];
  accum2(X   + base, Wr + o * 640 + 0,   Wu + o * 640 + 0,   ar, az);
  accum2(H1f + base, Wr + o * 640 + 128, Wu + o * 640 + 128, ar, az);
  accum2(H2f + base, Wr + o * 640 + 256, Wu + o * 640 + 256, ar, az);
  accum2(H1b + base, Wr + o * 640 + 384, Wu + o * 640 + 384, ar, az);
  accum2(H2b + base, Wr + o * 640 + 512, Wu + o * 640 + 512, ar, az);
  size_t oi = (size_t)(b * HH + o) * NN + n;
  float rv = 1.f / (1.f + __expf(-ar));
  z[oi] = 1.f / (1.f + __expf(-az));
  Xc[(size_t)(b * 128 + 64 + o) * NN + n] = rv * h[oi];
}

// ---------------------------------------------------------------------------
// candidate + GRU update: cst = tanh(Wc*cat_c + bc); h = z*h + (1-z)*cst.
// Writes new h into X's h-rows so X becomes the decoder input [ins;h].
// ---------------------------------------------------------------------------
__global__ __launch_bounds__(256) void k_cand(const float* __restrict__ Xc,
                                              const float* __restrict__ C1f,
                                              const float* __restrict__ C2f,
                                              const float* __restrict__ C1b,
                                              const float* __restrict__ C2b,
                                              const float* __restrict__ Wc,
                                              const float* __restrict__ bc,
                                              const float* __restrict__ z,
                                              float* __restrict__ h,
                                              float* __restrict__ X) {
  int idx = blockIdx.x * 256 + threadIdx.x;   // HH*BB*NN
  if (idx >= HH * BB * NN) return;
  int o   = idx / (BB * NN);
  int rem = idx % (BB * NN);
  int b = rem / NN, n = rem % NN;
  size_t base = (size_t)b * 128 * NN + n;
  float ac = bc[o];
  accum1(Xc  + base, Wc + o * 640 + 0,   ac);
  accum1(C1f + base, Wc + o * 640 + 128, ac);
  accum1(C2f + base, Wc + o * 640 + 256, ac);
  accum1(C1b + base, Wc + o * 640 + 384, ac);
  accum1(C2b + base, Wc + o * 640 + 512, ac);
  float cst = tanhf(ac);
  size_t hi = (size_t)(b * HH + o) * NN + n;
  float zz = z[hi];
  float hn = zz * h[hi] + (1.f - zz) * cst;
  h[hi] = hn;
  X[(size_t)(b * 128 + 64 + o) * NN + n] = hn;
}

// ---------------------------------------------------------------------------
// Fused decoder: O1 = Wdec*[D1f;D1b]+bdec ; O2 = prelu(Wlin*[O1;h]+blin) ;
// xs2 = Wread*[O2;h]+bread ; xs1 = Wf*h+bf.  LDS-staged across the two
// cross-channel stages. Block = 32 nodes x 64 channels (8 per thread).
// ---------------------------------------------------------------------------
__global__ __launch_bounds__(256) void k_decoder(
    const float* __restrict__ D1f, const float* __restrict__ D1b,
    const float* __restrict__ h,
    const float* __restrict__ Wdec, const float* __restrict__ bdec,
    const float* __restrict__ Wlin, const float* __restrict__ blin,
    const float* __restrict__ Wread, const float* __restrict__ bread,
    const float* __restrict__ Wf, const float* __restrict__ bf,
    const float* __restrict__ prelu_a,
    float* __restrict__ out_imp, float* __restrict__ out_pred,
    float* __restrict__ pred, int t) {
  __shared__ float sO1[HH][33];
  __shared__ float sO2[HH][33];
  int tid = threadIdx.x;
  int nn  = tid & 31;           // node within tile
  int og  = tid >> 5;           // channel group (8 channels each)
  int blk = blockIdx.x;         // BB * NN/32 = 64 blocks
  int b     = blk >> 5;
  int ntile = blk & 31;
  int n = ntile * 32 + nn;

  float acc[8];
  // stage 1: O1[o][n]
#pragma unroll
  for (int j = 0; j < 8; ++j) acc[j] = bdec[og * 8 + j];
  for (int cc = 0; cc < 128; ++cc) {
    float vf = D1f[(size_t)(b * 128 + cc) * NN + n];
    float vb = D1b[(size_t)(b * 128 + cc) * NN + n];
#pragma unroll
    for (int j = 0; j < 8; ++j) {
      int o = og * 8 + j;
      acc[j] = fmaf(Wdec[o * 256 + cc], vf, acc[j]);
      acc[j] = fmaf(Wdec[o * 256 + 128 + cc], vb, acc[j]);
    }
  }
#pragma unroll
  for (int j = 0; j < 8; ++j) sO1[og * 8 + j][nn] = acc[j];
  __syncthreads();

  // stage 2: O2 = prelu(Wlin*[O1;h] + blin)
  float a = prelu_a[0];
#pragma unroll
  for (int j = 0; j < 8; ++j) acc[j] = blin[og * 8 + j];
  for (int cc = 0; cc < HH; ++cc) {
    float v1 = sO1[cc][nn];
    float vh = h[(size_t)(b * HH + cc) * NN + n];
#pragma unroll
    for (int j = 0; j < 8; ++j) {
      int o = og * 8 + j;
      acc[j] = fmaf(Wlin[o * 128 + cc], v1, acc[j]);
      acc[j] = fmaf(Wlin[o * 128 + 64 + cc], vh, acc[j]);
    }
  }
#pragma unroll
  for (int j = 0; j < 8; ++j) {
    float s = acc[j];
    sO2[og * 8 + j][nn] = (s >= 0.f) ? s : a * s;
  }
  __syncthreads();

  // stage 3: readouts (one thread per node)
  if (og == 0) {
    float s2 = bread[0];
    float s1 = bf[0];
    for (int cc = 0; cc < HH; ++cc) {
      float vh = h[(size_t)(b * HH + cc) * NN + n];
      s2 = fmaf(Wread[cc], sO2[cc][nn], s2);
      s2 = fmaf(Wread[64 + cc], vh, s2);
      s1 = fmaf(Wf[cc], vh, s1);
    }
    out_imp[(size_t)(b * NN + n) * TT + t]  = s2;
    out_pred[(size_t)(b * NN + n) * TT + t] = s1;
    pred[b * NN + n] = s2;
  }
}

// ---------------------------------------------------------------------------
extern "C" void kernel_launch(void* const* d_in, const int* in_sizes, int n_in,
                              void* d_out, int out_size, void* d_ws, size_t ws_size,
                              hipStream_t stream) {
  (void)in_sizes; (void)n_in; (void)out_size; (void)ws_size;
  const float*         x     = (const float*)d_in[0];
  const float*         adj   = (const float*)d_in[1];
  const unsigned char* mask  = (const unsigned char*)d_in[2];
  const float*         u     = (const float*)d_in[3];
  const float* Wp = (const float*)d_in[4];  const float* bp = (const float*)d_in[5];
  const float* Wr = (const float*)d_in[6];  const float* br = (const float*)d_in[7];
  const float* Wu = (const float*)d_in[8];  const float* bu = (const float*)d_in[9];
  const float* Wc = (const float*)d_in[10]; const float* bc = (const float*)d_in[11];
  const float* Wf = (const float*)d_in[12]; const float* bf = (const float*)d_in[13];
  const float* Wdec = (const float*)d_in[14]; const float* bdec = (const float*)d_in[15];
  const float* Wlin = (const float*)d_in[16]; const float* blin = (const float*)d_in[17];
  const float* Wread = (const float*)d_in[18]; const float* bread = (const float*)d_in[19];
  const float* prelu_a = (const float*)d_in[20];
  float* out = (float*)d_out;
  float* out_imp  = out;                          // (B,1,N,T)
  float* out_pred = out + (size_t)BB * NN * TT;   // (B,1,N,T)

  // workspace carve-up
  char* wsb = (char*)d_ws;
  size_t off = 0;
  auto carve = [&](size_t bytes) -> void* {
    void* p = wsb + off;
    off = (off + bytes + 255) & ~(size_t)255;
    return p;
  };
  _Float16* Sf   = (_Float16*)carve((size_t)NN * NN * 2);
  _Float16* Sb   = (_Float16*)carve((size_t)NN * NN * 2);
  float* invr    = (float*)carve(NN * 4);
  float* invc    = (float*)carve(NN * 4);
  float* X       = (float*)carve((size_t)MROWS * NN * 4);
  float* Xc      = (float*)carve((size_t)MROWS * NN * 4);
  float* H1f     = (float*)carve((size_t)MROWS * NN * 4);
  float* H2f     = (float*)carve((size_t)MROWS * NN * 4);
  float* H1b     = (float*)carve((size_t)MROWS * NN * 4);
  float* H2b     = (float*)carve((size_t)MROWS * NN * 4);
  float* zbuf    = (float*)carve((size_t)BB * HH * NN * 4);
  float* hbuf    = (float*)carve((size_t)BB * HH * NN * 4);
  float* predbuf = (float*)carve((size_t)BB * NN * 4);

  const int BN_BLK   = (BB * NN + 255) / 256;        // 8
  const int OBN_BLK  = (HH * BB * NN + 255) / 256;   // 512
  const int HOP2_BLK = 64;                           // 512 waves (both supports)
  const int DEC_BLK  = BB * (NN / 32);               // 64

  // deterministic init of recurrent state
  k_zero<<<(BB * HH * NN + 255) / 256, 256, 0, stream>>>(hbuf, BB * HH * NN);
  k_zero<<<BN_BLK, 256, 0, stream>>>(predbuf, BB * NN);

  // one-time support prep (f16 B-fragment-swizzled transposed supports)
  k_sums<<<NN, 256, 0, stream>>>(adj, invr, invc);
  k_swz<<<(NN * NN) / 256, 256, 0, stream>>>(adj, invr, invc, Sf, Sb);

  for (int t = 0; t < TT; ++t) {
    // input projection + xg = [ins ; h], Xc's ins rows
    k_proj<<<BN_BLK, 256, 0, stream>>>(x, mask, u, Wp, bp, predbuf, hbuf, X, Xc, t);

    // gate-path diffusion: both supports per launch
    k_hop2<<<HOP2_BLK, 256, 0, stream>>>(X,   Sf, H1f, X,   Sb, H1b);
    k_hop2<<<HOP2_BLK, 256, 0, stream>>>(H1f, Sf, H2f, H1b, Sb, H2b);
    k_gates<<<OBN_BLK, 256, 0, stream>>>(X, H1f, H2f, H1b, H2b, Wr, br, Wu, bu,
                                         hbuf, zbuf, Xc);

    // candidate-path diffusion on [ins ; r*h] (reuse hop buffers)
    k_hop2<<<HOP2_BLK, 256, 0, stream>>>(Xc,  Sf, H1f, Xc,  Sb, H1b);
    k_hop2<<<HOP2_BLK, 256, 0, stream>>>(H1f, Sf, H2f, H1b, Sb, H2b);
    k_cand<<<OBN_BLK, 256, 0, stream>>>(Xc, H1f, H2f, H1b, H2b, Wc, bc, zbuf, hbuf, X);

    // decoder: order-1, no self -> 2 hops on [ins ; h_new] in one launch
    k_hop2<<<HOP2_BLK, 256, 0, stream>>>(X, Sf, H1f, X, Sb, H1b);
    k_decoder<<<DEC_BLK, 256, 0, stream>>>(H1f, H1b, hbuf, Wdec, bdec, Wlin, blin,
                                           Wread, bread, Wf, bf, prelu_a,
                                           out_imp, out_pred, predbuf, t);
  }
}
